// T_TGCN_18485539242711
// MI455X (gfx1250) — matrix operand order repaired and checked by reading, not compiled, not run on hardware
//
#include <hip/hip_runtime.h>
#include <math.h>

#define NNODES 50000
#define FIN    128
#define FOUT   128
#define F3     384      // 3 * FOUT, merged conv output width
#define KGATE  256      // concat(conv, h) width
#define DD     7

typedef __attribute__((ext_vector_type(16))) __bf16 v16bf;
typedef __attribute__((ext_vector_type(8)))  float  v8f;

// ---------------------------------------------------------------------------
// WMMA fragment loaders (wave32, 16-bit element layouts per CDNA5 ISA 7.12.2)
// A (16x32, MxK): lane 0-15 -> row m0+lane, K = k0+[0..7] and k0+16+[0..7]
//                 lane16-31 -> row m0+lane-16, K = k0+8+[0..7], k0+24+[0..7]
// ---------------------------------------------------------------------------
static __device__ __forceinline__ v16bf
load_afrag_bf16(const __bf16* A, int ld, int m0, int k0, int lane) {
  int m  = lane & 15;
  int kb = k0 + ((lane >> 4) << 3);
  const __bf16* p = A + (size_t)(m0 + m) * ld + kb;
  union { v16bf v; uint4 q[2]; } u;
  u.q[0] = *(const uint4*)(p);
  u.q[1] = *(const uint4*)(p + 16);
  return u.v;
}

// B (32x16, KxN), weights stored transposed [outCol][K]:
// lane 0-15 -> col n0+lane, K = k0+[0..7], k0+16+[0..7] ; lane16-31 -> K = k0+8+[0..7], k0+24+[0..7]
static __device__ __forceinline__ v16bf
load_bfrag_bf16(const __bf16* Wt, int ld, int n0, int k0, int lane) {
  int n = lane & 15;
  const __bf16* p = Wt + (size_t)(n0 + n) * ld + k0 + ((lane >> 4) << 3);
  union { v16bf v; uint4 q[2]; } u;
  u.q[0] = *(const uint4*)(p);
  u.q[1] = *(const uint4*)(p + 16);
  return u.v;
}

// A-fragment from a conceptual f32 [N, 256] = [A1(128 cols) | A2(128 cols)],
// converted to bf16 on the fly. k-segments never straddle the 128 boundary
// because steps are 32-aligned.
static __device__ __forceinline__ v16bf
load_afrag_f32cat(const float* A1, int lda1, const float* A2, int lda2,
                  int m0, int k0, int lane) {
  int m   = lane & 15;
  int kb0 = k0 + ((lane >> 4) << 3);
  union { v16bf v; __bf16 e[16]; } u;
#pragma unroll
  for (int part = 0; part < 2; ++part) {
    int kb = kb0 + part * 16;
    const float* src = (kb < FOUT) ? (A1 + (size_t)(m0 + m) * lda1 + kb)
                                   : (A2 + (size_t)(m0 + m) * lda2 + (kb - FOUT));
    float4 f0 = *(const float4*)(src);
    float4 f1 = *(const float4*)(src + 4);
    int o = part * 8;
    u.e[o+0] = (__bf16)f0.x; u.e[o+1] = (__bf16)f0.y;
    u.e[o+2] = (__bf16)f0.z; u.e[o+3] = (__bf16)f0.w;
    u.e[o+4] = (__bf16)f1.x; u.e[o+5] = (__bf16)f1.y;
    u.e[o+6] = (__bf16)f1.z; u.e[o+7] = (__bf16)f1.w;
  }
  return u.v;
}

// ---------------------------------------------------------------------------
// Normalization prep
// ---------------------------------------------------------------------------
__global__ void k_init_deg(float* deg, int n) {
  int i = blockIdx.x * blockDim.x + threadIdx.x;
  if (i < n) deg[i] = 2.0f;                 // improved=True self-loop weight
}

__global__ void k_deg_scatter(const int* ei, const float* w, float* deg, int E) {
  int e = blockIdx.x * blockDim.x + threadIdx.x;
  if (e < E) {
    int c = ei[E + e];
    __hip_atomic_fetch_add(&deg[c], w[e], __ATOMIC_RELAXED, __HIP_MEMORY_SCOPE_AGENT);
  }
}

__global__ void k_dis(float* deg, int n) {   // in place deg -> deg^-1/2
  int i = blockIdx.x * blockDim.x + threadIdx.x;
  if (i < n) { float d = deg[i]; deg[i] = (d > 0.0f) ? rsqrtf(d) : 0.0f; }
}

// ---------------------------------------------------------------------------
// bf16 conversions / weight transposes
// ---------------------------------------------------------------------------
__global__ void k_cvt_x(const float* x, __bf16* xb, int n) {
  int i = blockIdx.x * blockDim.x + threadIdx.x;
  if (i < n) xb[i] = (__bf16)x[i];
}

__global__ void k_prep_w(const float* Wz, const float* Wr, const float* Wh,
                         __bf16* Wcat_t,
                         const float* Lz, const float* Lr, const float* Lh,
                         __bf16* Lzt, __bf16* Lrt, __bf16* Lht) {
  int idx = blockIdx.x * blockDim.x + threadIdx.x;
  const int NW = F3 * FIN;                    // 49152: Wcat_t[384][128]
  if (idx < NW) {
    int o = idx / FIN, k = idx % FIN;
    float v = (o < FOUT)     ? Wz[k * FOUT + o]
            : (o < 2 * FOUT) ? Wr[k * FOUT + (o - FOUT)]
                             : Wh[k * FOUT + (o - 2 * FOUT)];
    Wcat_t[(size_t)o * FIN + k] = (__bf16)v;
  } else if (idx < NW + 3 * FOUT * KGATE) {   // Lt[128][256] each
    int j = idx - NW;
    int which = j / (FOUT * KGATE);
    int rem   = j % (FOUT * KGATE);
    int o = rem / KGATE, k = rem % KGATE;
    const float* L = (which == 0) ? Lz : (which == 1) ? Lr : Lh;
    __bf16* Lt     = (which == 0) ? Lzt : (which == 1) ? Lrt : Lht;
    Lt[(size_t)o * KGATE + k] = (__bf16)L[(size_t)k * FOUT + o];
  }
}

// ---------------------------------------------------------------------------
// GEMM 1: xw[N,384] = x[N,128] @ [Wz|Wr|Wh]  (bf16 WMMA, f32 accum)
// ---------------------------------------------------------------------------
__global__ void k_gemm_xw(const __bf16* xb, const __bf16* Wt, float* xw, int Mtiles) {
  int wid = blockIdx.x * (blockDim.x >> 5) + (threadIdx.x >> 5);
  const int NT = F3 / 16;                     // 24
  if (wid >= Mtiles * NT) return;             // wave-uniform: EXEC stays all-1s
  int mt = wid / NT, nt = wid % NT;
  int lane = threadIdx.x & 31;
  v8f c = {};
#pragma unroll
  for (int k0 = 0; k0 < FIN; k0 += 32) {
    v16bf a = load_afrag_bf16(xb, FIN, mt * 16, k0, lane);
    v16bf b = load_bfrag_bf16(Wt, FIN, nt * 16, k0, lane);
    c = __builtin_amdgcn_wmma_f32_16x16x32_bf16(false, a, false, b, (short)0, c, false, false);
  }
  int n  = nt * 16 + (lane & 15);
  int mb = mt * 16 + ((lane >> 4) ? 8 : 0);
#pragma unroll
  for (int r = 0; r < 8; ++r)
    xw[(size_t)(mb + r) * F3 + n] = c[r];
}

// ---------------------------------------------------------------------------
// agg init: bias + self-loop contribution (norm_self = dis^2 * 2.0)
// ---------------------------------------------------------------------------
__global__ void k_init_agg(const float* xw, const float* dis,
                           const float* bz, const float* br, const float* bh,
                           float* agg, int n) {
  int i = blockIdx.x * blockDim.x + threadIdx.x;
  if (i >= n) return;
  int node = i / F3, f = i % F3;
  float d = dis[node];
  float b = (f < FOUT) ? bz[f] : (f < 2 * FOUT) ? br[f - FOUT] : bh[f - 2 * FOUT];
  agg[i] = b + (2.0f * d * d) * xw[i];
}

// one wave per edge: 12 features per lane, f32 atomics into L2-resident agg
__global__ void k_edge_scatter(const int* ei, const float* w, const float* dis,
                               const float* xw, float* agg, int E) {
  int e = blockIdx.x * (blockDim.x >> 5) + (threadIdx.x >> 5);
  if (e >= E) return;
  int lane = threadIdx.x & 31;
  int r = ei[e], cdst = ei[E + e];
  float nrm = dis[r] * w[e] * dis[cdst];
  const float* src = xw + (size_t)r * F3;
  float*       dst = agg + (size_t)cdst * F3;
#pragma unroll
  for (int j = 0; j < F3 / 32; ++j) {
    int f = lane + j * 32;
    __hip_atomic_fetch_add(&dst[f], nrm * src[f], __ATOMIC_RELAXED, __HIP_MEMORY_SCOPE_AGENT);
  }
}

// ---------------------------------------------------------------------------
// GEMM 2: gate = act( [A1|A2][N,256] @ Lt^T + bias )   (bf16 WMMA)
// ACT: 0 none, 1 sigmoid, 2 tanh
// ---------------------------------------------------------------------------
template <int ACT>
__global__ void k_gate_gemm(const float* A1, int lda1, const float* A2, int lda2,
                            const __bf16* Wt, const float* bias, float* out, int Mtiles) {
  int wid = blockIdx.x * (blockDim.x >> 5) + (threadIdx.x >> 5);
  const int NT = FOUT / 16;                   // 8
  if (wid >= Mtiles * NT) return;
  int mt = wid / NT, nt = wid % NT;
  int lane = threadIdx.x & 31;
  v8f c = {};
#pragma unroll
  for (int k0 = 0; k0 < KGATE; k0 += 32) {
    v16bf a = load_afrag_f32cat(A1, lda1, A2, lda2, mt * 16, k0, lane);
    v16bf b = load_bfrag_bf16(Wt, KGATE, nt * 16, k0, lane);
    c = __builtin_amdgcn_wmma_f32_16x16x32_bf16(false, a, false, b, (short)0, c, false, false);
  }
  int n  = nt * 16 + (lane & 15);
  int mb = mt * 16 + ((lane >> 4) ? 8 : 0);
  float bv = bias[n];
#pragma unroll
  for (int r = 0; r < 8; ++r) {
    float v = c[r] + bv;
    if (ACT == 1)      v = 1.0f / (1.0f + __expf(-v));
    else if (ACT == 2) v = tanhf(v);
    out[(size_t)(mb + r) * FOUT + n] = v;
  }
}

// ---------------------------------------------------------------------------
// elementwise tail
// ---------------------------------------------------------------------------
__global__ void k_hr(const float* h0, float* R, int n) {   // R <- h0 * R
  int i = blockIdx.x * blockDim.x + threadIdx.x;
  if (i < n) R[i] = h0[i] * R[i];
}

__global__ void k_combine(const float* Z, const float* h0, const float* Ht,
                          float* H, int n) {               // H = Z*h0 + (1-Z)*Ht
  int i = blockIdx.x * blockDim.x + threadIdx.x;
  if (i < n) { float z = Z[i]; H[i] = z * h0[i] + (1.0f - z) * Ht[i]; }
}

__global__ void k_head(const float* H, const float* lw, const float* lb,
                       float* y, int n) {                  // y = relu(H) @ lin_w + b
  int i = blockIdx.x * blockDim.x + threadIdx.x;
  if (i >= n) return;
  int node = i / DD, d = i % DD;
  const float* h = H + (size_t)node * FOUT;
  float acc = lb[d];
#pragma unroll 4
  for (int f = 0; f < FOUT; ++f) {
    float v = h[f];
    v = v > 0.0f ? v : 0.0f;
    acc += v * lw[f * DD + d];
  }
  y[i] = acc;
}

// ---------------------------------------------------------------------------
extern "C" void kernel_launch(void* const* d_in, const int* in_sizes, int n_in,
                              void* d_out, int out_size, void* d_ws, size_t ws_size,
                              hipStream_t stream) {
  const float* x    = (const float*)d_in[0];
  const int*   ei   = (const int*)  d_in[1];
  const float* ew   = (const float*)d_in[2];
  const float* h0   = (const float*)d_in[3];
  const float* Wz   = (const float*)d_in[4];
  const float* bz   = (const float*)d_in[5];
  const float* Wr   = (const float*)d_in[6];
  const float* br   = (const float*)d_in[7];
  const float* Wh   = (const float*)d_in[8];
  const float* bh   = (const float*)d_in[9];
  const float* Lz_w = (const float*)d_in[10];
  const float* Lz_b = (const float*)d_in[11];
  const float* Lr_w = (const float*)d_in[12];
  const float* Lr_b = (const float*)d_in[13];
  const float* Lh_w = (const float*)d_in[14];
  const float* Lh_b = (const float*)d_in[15];
  const float* lw   = (const float*)d_in[16];
  const float* lb   = (const float*)d_in[17];

  const int N = in_sizes[0] / FIN;          // 50000 (divisible by 16)
  const int E = in_sizes[1] / 2;            // 800000

  // workspace carve-up (256B aligned)
  char* ws = (char*)d_ws;
  size_t off = 0;
  auto carve = [&](size_t bytes) { char* p = ws + off; off = (off + bytes + 255) & ~(size_t)255; return p; };
  float*  dis   = (float*) carve((size_t)N * 4);
  __bf16* xb    = (__bf16*)carve((size_t)N * FIN * 2);
  __bf16* Wcat  = (__bf16*)carve((size_t)F3 * FIN * 2);
  __bf16* Lzt   = (__bf16*)carve((size_t)FOUT * KGATE * 2);
  __bf16* Lrt   = (__bf16*)carve((size_t)FOUT * KGATE * 2);
  __bf16* Lht   = (__bf16*)carve((size_t)FOUT * KGATE * 2);
  float*  xw    = (float*) carve((size_t)N * F3 * 4);
  float*  agg   = (float*) carve((size_t)N * F3 * 4);
  float*  Zg    = (float*) carve((size_t)N * FOUT * 4);
  float*  Rg    = (float*) carve((size_t)N * FOUT * 4);
  float*  Htl   = (float*) carve((size_t)N * FOUT * 4);

  float* H = (float*)d_out;                 // [N,128]
  float* y = H + (size_t)N * FOUT;          // [N,7]

  const int TB = 256;
  auto blocks = [](long long n, int tb) { return (int)((n + tb - 1) / tb); };

  // 1) degree / normalization
  k_init_deg   <<<blocks(N, TB), TB, 0, stream>>>(dis, N);
  k_deg_scatter<<<blocks(E, TB), TB, 0, stream>>>(ei, ew, dis, E);
  k_dis        <<<blocks(N, TB), TB, 0, stream>>>(dis, N);

  // 2) bf16 conversions + weight transposes
  k_cvt_x <<<blocks((long long)N * FIN, TB), TB, 0, stream>>>(x, xb, N * FIN);
  k_prep_w<<<blocks(F3 * FIN + 3 * FOUT * KGATE, TB), TB, 0, stream>>>(
      Wz, Wr, Wh, Wcat, Lz_w, Lr_w, Lh_w, Lzt, Lrt, Lht);

  // 3) xw = x @ [Wz|Wr|Wh]   (one 16x16 tile per wave, 8 waves/block)
  {
    int Mt = N / 16, tiles = Mt * (F3 / 16);
    k_gemm_xw<<<blocks(tiles, 8), TB, 0, stream>>>(xb, Wcat, xw, Mt);
  }

  // 4) aggregate: bias + self-loop init, then edge scatter (L2-resident atomics)
  k_init_agg<<<blocks((long long)N * F3, TB), TB, 0, stream>>>(xw, dis, bz, br, bh, agg,
                                                               N * F3);
  k_edge_scatter<<<blocks(E, 8), TB, 0, stream>>>(ei, ew, dis, xw, agg, E);

  // 5) GRU gates via WMMA GEMMs over concat inputs
  {
    int Mt = N / 16, tiles = Mt * (FOUT / 16);
    k_gate_gemm<1><<<blocks(tiles, 8), TB, 0, stream>>>(agg + 0,   F3, h0, FOUT, Lzt, Lz_b, Zg, Mt);
    k_gate_gemm<1><<<blocks(tiles, 8), TB, 0, stream>>>(agg + 128, F3, h0, FOUT, Lrt, Lr_b, Rg, Mt);
    k_hr<<<blocks((long long)N * FOUT, TB), TB, 0, stream>>>(h0, Rg, N * FOUT);
    k_gate_gemm<2><<<blocks(tiles, 8), TB, 0, stream>>>(agg + 256, F3, Rg, FOUT, Lht, Lh_b, Htl, Mt);
  }

  // 6) H = Z*h0 + (1-Z)*H~ ; y = relu(H) @ lin_w + lin_b
  k_combine<<<blocks((long long)N * FOUT, TB), TB, 0, stream>>>(Zg, h0, Htl, H, N * FOUT);
  k_head   <<<blocks((long long)N * DD, TB), TB, 0, stream>>>(H, lw, lb, y, N * DD);
}
